// lstm_seq2seq_8572754722968
// MI455X (gfx1250) — matrix operand order repaired and checked by reading, not compiled, not run on hardware
//
#include <hip/hip_runtime.h>

typedef __attribute__((ext_vector_type(16))) _Float16 v16h;
typedef __attribute__((ext_vector_type(8)))  _Float16 v8h;
typedef __attribute__((ext_vector_type(8)))  float    v8f;
typedef __attribute__((ext_vector_type(4)))  float    v4f;

#define HID   64
#define BATCH 1024
#define T_ENC 1000
#define T_DEC 1000

__device__ __forceinline__ float ftanh(float x) {
#if __has_builtin(__builtin_amdgcn_tanhf)
    return __builtin_amdgcn_tanhf(x);          // native v_tanh_f32 (1 TRANS op)
#else
    return 1.0f - 2.0f * __builtin_amdgcn_rcpf(__expf(2.0f * x) + 1.0f);
#endif
}
__device__ __forceinline__ float fsig(float x) {
    // sigmoid(x) = 0.5 + 0.5*tanh(0.5*x): 1 TRANS + 1 mul + 1 fma
    return __builtin_fmaf(0.5f, ftanh(0.5f * x), 0.5f);
}

// B-matrix fragment for v_wmma_f32_16x16x32_f16.
// Layout: lane L -> N = L%16, K = (L/16)*16 + e  (e = 0..15, consecutive).
// B[k][n] = Whh[j=n_global][k]  (gates = h @ Whh^T), Whh row-major (256,64).
__device__ __forceinline__ v16h load_b_frag(const float* __restrict__ Whh,
                                            int j, int kb) {
    const v4f* s = (const v4f*)(Whh + j * HID + kb);
    v4f w0 = s[0], w1 = s[1], w2 = s[2], w3 = s[3];
    v16h f;
#pragma unroll
    for (int e = 0; e < 4; e++) {
        f[e]      = (_Float16)w0[e];
        f[4 + e]  = (_Float16)w1[e];
        f[8 + e]  = (_Float16)w2[e];
        f[12 + e] = (_Float16)w3[e];
    }
    return f;
}

// A-matrix fragment from LDS h buffer (f16 [16 batch][64 hid], row stride 128B).
// Layout: lane L -> M = L%16; elements 0..15 -> K = k0+half*16+e (contiguous,
// matching the B fragments). Two 16B LDS reads per fragment.
__device__ __forceinline__ v16h load_a_frag(const _Float16* __restrict__ hb,
                                            int m, int half, int k0) {
    const _Float16* row = hb + m * HID;
    int ka = k0 + half * 16;
    v8h lo = *(const v8h*)(row + ka);
    v8h hi = *(const v8h*)(row + ka + 8);
    v16h a;
#pragma unroll
    for (int e = 0; e < 8; e++) { a[e] = lo[e]; a[8 + e] = hi[e]; }
    return a;
}

template <bool DEC>
__device__ __forceinline__ void lstm_phase(
    const float* __restrict__ Wih, const float* __restrict__ Whh,
    const float* __restrict__ bih, const float* __restrict__ bhh,
    const float* __restrict__ xseq, int T,
    float* __restrict__ out, const float* __restrict__ linW, float linb,
    v8f& c, int& par,
    _Float16* __restrict__ hbuf,   // [2][16*HID] f16, double-buffered
    int lane, int wave, int bbase) {

    const int l16  = lane & 15;
    const int half = lane >> 4;
    const int ncol = wave * 16 + l16;   // hidden unit (N column) of this lane

    // Per-lane loop invariants: combined bias and Wih column per gate (IN==1).
    float bias[4], wih[4];
#pragma unroll
    for (int g = 0; g < 4; g++) {
        int j   = g * HID + ncol;
        bias[g] = bih[j] + bhh[j];
        wih[g]  = Wih[j];
    }

    // Pin Whh fragments in VGPRs for the whole scan: 4 gates x 2 K-halves.
    v16h Bf[4][2];
#pragma unroll
    for (int g = 0; g < 4; g++) {
        int j = g * HID + ncol;
        Bf[g][0] = load_b_frag(Whh, j, 0 + half * 16);
        Bf[g][1] = load_b_frag(Whh, j, 32 + half * 16);
    }

    // Projection B fragments: B[k][0] = linW[k], all other columns zero.
    // out[t] = h_t @ linW + linb is a 16x64x1 GEMM; wave 3 computes it from the
    // A fragments it reloads anyway at step t+1 (2 extra WMMAs, no shuffles).
    v16h Bl0, Bl1;
    if (DEC) {
#pragma unroll
        for (int e = 0; e < 16; e++) {
            float w0 = (l16 == 0) ? linW[half * 16 + e] : 0.0f;
            float w1 = (l16 == 0) ? linW[32 + half * 16 + e] : 0.0f;
            Bl0[e] = (_Float16)w0;
            Bl1[e] = (_Float16)w1;
        }
    }

    for (int t = 0; t < T; t++) {
        const _Float16* hcur  = hbuf + par * (16 * HID);
        _Float16*       hnext = hbuf + (par ^ 1) * (16 * HID);

        v16h a0 = load_a_frag(hcur, l16, half, 0);
        v16h a1 = load_a_frag(hcur, l16, half, 32);

        // wave 3 emits out[t-1] from h_{t-1} (the A fragments just loaded)
        if (DEC && wave == 3 && t >= 1) {
            v8f pacc;
#pragma unroll
            for (int e = 0; e < 8; e++) pacc[e] = linb;
            pacc = __builtin_amdgcn_wmma_f32_16x16x32_f16(
                false, a0, false, Bl0, (short)0, pacc, false, false);
            pacc = __builtin_amdgcn_wmma_f32_16x16x32_f16(
                false, a1, false, Bl1, (short)0, pacc, false, false);
            if (l16 == 0) {   // lanes 0 & 16 hold D[:,0]: b = e + 8*half
                float* op = out + (size_t)(t - 1) * BATCH + bbase + 8 * half;
#pragma unroll
                for (int e = 0; e < 8; e++) op[e] = pacc[e];
            }
        }

        // x for this tile: 8 batch rows per half (matches C-layout M = e + 8*half)
        float xv[8];
        if (DEC && t == 0) {
#pragma unroll
            for (int e = 0; e < 8; e++) xv[e] = 0.0f;
        } else {
            const float* xr = xseq + (size_t)(DEC ? (t - 1) : t) * BATCH + bbase + half * 8;
            v4f x0 = *(const v4f*)xr;
            v4f x1 = *(const v4f*)(xr + 4);
#pragma unroll
            for (int e = 0; e < 4; e++) { xv[e] = x0[e]; xv[4 + e] = x1[e]; }
        }

        // gates = bias + x*Wih + h @ Whh^T   (C pre-initialized, 2 WMMAs per gate)
        v8f acc[4];
#pragma unroll
        for (int g = 0; g < 4; g++) {
#pragma unroll
            for (int e = 0; e < 8; e++) acc[g][e] = __builtin_fmaf(xv[e], wih[g], bias[g]);
            acc[g] = __builtin_amdgcn_wmma_f32_16x16x32_f16(
                false, a0, false, Bf[g][0], (short)0, acc[g], false, false);
            acc[g] = __builtin_amdgcn_wmma_f32_16x16x32_f16(
                false, a1, false, Bf[g][1], (short)0, acc[g], false, false);
        }

        // i,f,g,o -> c,h update (PyTorch gate order), all wave-local
        v8f hv;
#pragma unroll
        for (int e = 0; e < 8; e++) {
            float ig = fsig(acc[0][e]);
            float fg = fsig(acc[1][e]);
            float gg = ftanh(acc[2][e]);
            float og = fsig(acc[3][e]);
            float cv = __builtin_fmaf(fg, c[e], ig * gg);
            c[e]  = cv;
            hv[e] = og * ftanh(cv);
        }

        // publish h_t (f16) into the other parity buffer
#pragma unroll
        for (int e = 0; e < 8; e++) {
            int b = e + 8 * half;
            hnext[b * HID + ncol] = (_Float16)hv[e];
        }

        __syncthreads();   // one barrier per step (parity double-buffering)
        par ^= 1;
    }

    // epilogue: out[T-1] from the final h (now in hbuf[par])
    if (DEC && wave == 3) {
        const _Float16* hcur = hbuf + par * (16 * HID);
        v16h a0 = load_a_frag(hcur, l16, half, 0);
        v16h a1 = load_a_frag(hcur, l16, half, 32);
        v8f pacc;
#pragma unroll
        for (int e = 0; e < 8; e++) pacc[e] = linb;
        pacc = __builtin_amdgcn_wmma_f32_16x16x32_f16(
            false, a0, false, Bl0, (short)0, pacc, false, false);
        pacc = __builtin_amdgcn_wmma_f32_16x16x32_f16(
            false, a1, false, Bl1, (short)0, pacc, false, false);
        if (l16 == 0) {
            float* op = out + (size_t)(T - 1) * BATCH + bbase + 8 * half;
#pragma unroll
            for (int e = 0; e < 8; e++) op[e] = pacc[e];
        }
    }
}

__global__ __launch_bounds__(128) void lstm_seq2seq_kernel(
    const float* __restrict__ input_batch, const float* __restrict__ target_batch,
    const float* __restrict__ eWih, const float* __restrict__ eWhh,
    const float* __restrict__ ebih, const float* __restrict__ ebhh,
    const float* __restrict__ dWih, const float* __restrict__ dWhh,
    const float* __restrict__ dbih, const float* __restrict__ dbhh,
    const float* __restrict__ linW, const float* __restrict__ linb,
    float* __restrict__ out) {

    __shared__ __align__(16) _Float16 hbuf[2 * 16 * HID];

    const int tid   = threadIdx.x;
    const int lane  = tid & 31;
    const int wave  = tid >> 5;
    const int bbase = blockIdx.x * 16;

    for (int i = tid; i < 2 * 16 * HID; i += 128) hbuf[i] = (_Float16)0.0f;
    __syncthreads();

    v8f c;
#pragma unroll
    for (int e = 0; e < 8; e++) c[e] = 0.0f;
    int par = 0;

    // encoder: only final (h,c) matters; they stay live in LDS/regs
    lstm_phase<false>(eWih, eWhh, ebih, ebhh, input_batch, T_ENC,
                      nullptr, linW, 0.0f, c, par, hbuf,
                      lane, wave, bbase);
    // decoder (teacher forced) + per-step WMMA linear projection
    lstm_phase<true>(dWih, dWhh, dbih, dbhh, target_batch, T_DEC,
                     out, linW, linb[0], c, par, hbuf,
                     lane, wave, bbase);
}

extern "C" void kernel_launch(void* const* d_in, const int* in_sizes, int n_in,
                              void* d_out, int out_size, void* d_ws, size_t ws_size,
                              hipStream_t stream) {
    (void)in_sizes; (void)n_in; (void)out_size; (void)d_ws; (void)ws_size;
    const float* input_batch  = (const float*)d_in[0];
    const float* target_batch = (const float*)d_in[1];
    const float* eWih = (const float*)d_in[2];
    const float* eWhh = (const float*)d_in[3];
    const float* ebih = (const float*)d_in[4];
    const float* ebhh = (const float*)d_in[5];
    const float* dWih = (const float*)d_in[6];
    const float* dWhh = (const float*)d_in[7];
    const float* dbih = (const float*)d_in[8];
    const float* dbhh = (const float*)d_in[9];
    const float* linW = (const float*)d_in[10];
    const float* linb = (const float*)d_in[11];
    float* out = (float*)d_out;

    dim3 grid(BATCH / 16);   // 64 independent batch tiles
    dim3 block(128);         // 4 waves; wave w owns hidden units [16w,16w+16)
    lstm_seq2seq_kernel<<<grid, block, 0, stream>>>(
        input_batch, target_batch, eWih, eWhh, ebih, ebhh,
        dWih, dWhh, dbih, dbhh, linW, linb, out);
}
